// SimpleOptimizedSparseWindowAttention_4028679324324
// MI455X (gfx1250) — matrix operand order repaired and checked by reading, not compiled, not run on hardware
//
#include <hip/hip_runtime.h>
#include <hip/hip_bf16.h>

typedef __bf16 v16bf __attribute__((ext_vector_type(16)));
typedef __bf16 v8bf  __attribute__((ext_vector_type(8)));
typedef __bf16 v4bf  __attribute__((ext_vector_type(4)));
typedef float  v8f   __attribute__((ext_vector_type(8)));
typedef float  v4f   __attribute__((ext_vector_type(4)));

#define DIM   512
#define NHEAD 4
#define CH    128          // head dim
#define TT    8
#define HH    60
#define WW    108
#define NTOK  (TT*HH*WW)   // 51840
#define NWIN  144          // 12 x 12 windows of (5,9)
#define SEQ   360          // 8*5*9 tokens per window
#define SPAD  384          // padded to 24 tiles of 16
#define NJT   (SPAD/16)    // 24 key tiles
#define SCALE 0.08838834764831845f  // 128^-0.5

// ---------------------------------------------------------------- helpers ---

__device__ __forceinline__ v8f wmma_bf16(v16bf a, v16bf b, v8f c) {
  // D = A(16x32 bf16) * B(32x16 bf16) + C(16x16 f32)
  return __builtin_amdgcn_wmma_f32_16x16x32_bf16(false, a, false, b,
                                                 (short)0, c, false, false);
}

// A operand (16-bit 16x32): lane M = lane&15, g = lane>>4.
// halves 0..7  -> K =  8g .. 8g+7    (contiguous)
// halves 8..15 -> K = 16+8g .. 23+8g (contiguous)
__device__ __forceinline__ v16bf make_a(const __bf16* row, int g) {
  v8bf lo = *(const v8bf*)(row + 8 * g);
  v8bf hi = *(const v8bf*)(row + 16 + 8 * g);
  v16bf a = {};
#pragma unroll
  for (int h = 0; h < 8; ++h) { a[h] = lo[h]; a[h + 8] = hi[h]; }
  return a;
}

// B operand (16-bit 32x16): lane N = lane&15, g = lane>>4; halves 0..7 -> K 8g..8g+7, 8..15 -> K 16+8g..23+8g.
__device__ __forceinline__ v16bf make_b(const __bf16* row, int g) {
  v8bf lo = *(const v8bf*)(row + 8 * g);
  v8bf hi = *(const v8bf*)(row + 16 + 8 * g);
  v16bf b = {};
#pragma unroll
  for (int h = 0; h < 8; ++h) { b[h] = lo[h]; b[h + 8] = hi[h]; }
  return b;
}

// Async DMA: 16 bytes global -> LDS, tracked by ASYNCcnt (no VGPR data path).
__device__ __forceinline__ void async_load_b128(const __bf16* gptr, __bf16* lptr) {
  unsigned lds_off = (unsigned)(size_t)lptr;   // low 32 bits = LDS byte offset
  asm volatile("global_load_async_to_lds_b128 %0, %1, off"
               :: "v"(lds_off), "v"(gptr) : "memory");
}

// ds_swizzle xor reductions across the 16-lane groups of the C/D layout
template <int XORM>
__device__ __forceinline__ float swz(float v) {
  constexpr int imm = 0x1f | (XORM << 10);   // and=0x1f, or=0, xor=XORM
  int r = __builtin_amdgcn_ds_swizzle(__builtin_bit_cast(int, v), imm);
  return __builtin_bit_cast(float, r);
}
__device__ __forceinline__ float rowmax16(float v) {
  v = fmaxf(v, swz<1>(v)); v = fmaxf(v, swz<2>(v));
  v = fmaxf(v, swz<4>(v)); v = fmaxf(v, swz<8>(v));
  return v;
}
__device__ __forceinline__ float rowsum16(float v) {
  v += swz<1>(v); v += swz<2>(v); v += swz<4>(v); v += swz<8>(v);
  return v;
}

// token -> (window index, in-window sequence position)
__device__ __forceinline__ void tok2win(int token, int& win, int& s) {
  int t   = token / (HH * WW);
  int rem = token - t * (HH * WW);
  int h   = rem / WW;
  int w   = rem - h * WW;
  win = (h / 5) * 12 + (w / 9);
  s   = t * 45 + (h % 5) * 9 + (w % 9);
}

// ------------------------------------------------- kernel 1: QKV projection ---
// grid (NTOK/128, DIM/64, 3). Block = 128 tokens x 64 outputs, one matrix per z.
// Scatters bf16 results into per-(window,head) layout; V stored transposed.
__global__ __launch_bounds__(256) void k_qkv(
    const float* __restrict__ x,
    const float* __restrict__ Wq, const float* __restrict__ bq,
    const float* __restrict__ Wk, const float* __restrict__ bk,
    const float* __restrict__ Wv, const float* __restrict__ bv,
    __bf16* __restrict__ Qw, __bf16* __restrict__ Kw, __bf16* __restrict__ Vt)
{
  __shared__ __bf16 ldsX[128 * 32];
  __shared__ __bf16 ldsW[64 * 32];
  const int tid  = threadIdx.x;
  const int wave = tid >> 5, lane = tid & 31;
  const int g = lane >> 4, n = lane & 15;
  const int tok0 = blockIdx.x * 128;
  const int d0   = blockIdx.y * 64;
  const int z    = blockIdx.z;
  const float* Wm = (z == 0) ? Wq : (z == 1) ? Wk : Wv;
  const float* bm = (z == 0) ? bq : (z == 1) ? bk : bv;

  v8f acc[4] = {};
  for (int ks = 0; ks < DIM / 32; ++ks) {
    // stage 128x32 x-tile: float4 loads, 4-wide cvt, b64 LDS stores
#pragma unroll
    for (int i = 0; i < 4; ++i) {
      int c = i * 256 + tid;                   // 1024 float4 chunks
      int r = c >> 3, k4 = (c & 7) * 4;
      v4f f = *(const v4f*)&x[(size_t)(tok0 + r) * DIM + ks * 32 + k4];
      v4bf b;
#pragma unroll
      for (int h = 0; h < 4; ++h) b[h] = (__bf16)f[h];
      *(v4bf*)&ldsX[r * 32 + k4] = b;
    }
    // stage 64x32 weight tile
#pragma unroll
    for (int i = 0; i < 2; ++i) {
      int c = i * 256 + tid;                   // 512 float4 chunks
      int r = c >> 3, k4 = (c & 7) * 4;
      v4f f = *(const v4f*)&Wm[(size_t)(d0 + r) * DIM + ks * 32 + k4];
      v4bf b;
#pragma unroll
      for (int h = 0; h < 4; ++h) b[h] = (__bf16)f[h];
      *(v4bf*)&ldsW[r * 32 + k4] = b;
    }
    __syncthreads();
    v16bf a = make_a(&ldsX[(wave * 16 + n) * 32], g);
#pragma unroll
    for (int nt = 0; nt < 4; ++nt) {
      v16bf b = make_b(&ldsW[(nt * 16 + n) * 32], g);
      acc[nt] = wmma_bf16(a, b, acc[nt]);
    }
    __syncthreads();
  }

  // epilogue: bias + scatter into window layout (tok2win hoisted per row)
  float bb[4];
  int heads[4], cs[4];
#pragma unroll
  for (int nt = 0; nt < 4; ++nt) {
    int d = d0 + nt * 16 + n;
    bb[nt] = bm[d];
    heads[nt] = d >> 7;
    cs[nt] = d & 127;
  }
#pragma unroll
  for (int r = 0; r < 8; ++r) {
    int token = tok0 + wave * 16 + r + 8 * g;
    int win, s; tok2win(token, win, s);
#pragma unroll
    for (int nt = 0; nt < 4; ++nt) {
      float val = acc[nt][r] + bb[nt];
      if (z == 0)
        Qw[(size_t)((win * NHEAD + heads[nt]) * SPAD + s) * CH + cs[nt]] = (__bf16)val;
      else if (z == 1)
        Kw[(size_t)((win * NHEAD + heads[nt]) * SPAD + s) * CH + cs[nt]] = (__bf16)val;
      else
        Vt[(size_t)((win * NHEAD + heads[nt]) * CH + cs[nt]) * SPAD + s] = (__bf16)val;
    }
  }
}

// --------------------------------------------- kernel 2: flash attention ---
// grid (NWIN*NHEAD, SPAD/128). Each wave owns 16 query rows, Q in registers.
// K tiles (16x128) and V pair-tiles (128x32) are DMA'd into double-buffered
// LDS with GLOBAL_LOAD_ASYNC_TO_LDS_B128, pipelined one tile ahead.
// Per-thread async issue order: K0,V0,V0 | K1 | K2,V1,V1 | K3 | ... so after
// issuing this iteration's prefetches, waiting ASYNCcnt<=3 guarantees the
// current K tile (and any V pair issued before it) has landed.
__global__ __launch_bounds__(256) void k_attn(
    const __bf16* __restrict__ Qw, const __bf16* __restrict__ Kw,
    const __bf16* __restrict__ Vt, __bf16* __restrict__ AO)
{
  __shared__ __bf16 ldsK[2][16 * 128];          // double-buffered K tile
  __shared__ __bf16 ldsV[2][128 * 32];          // double-buffered V pair-tile
  __shared__ __bf16 ldsP[8][2][256];            // per-wave P tiles (layout xpose)
  const int tid  = threadIdx.x;
  const int wave = tid >> 5, lane = tid & 31;
  const int g = lane >> 4, n = lane & 15;
  const int wh    = blockIdx.x;
  const int qbase = blockIdx.y * 128 + wave * 16;

  const __bf16* Qp = Qw + (size_t)wh * SPAD * CH;
  const __bf16* Kp = Kw + (size_t)wh * SPAD * CH;
  const __bf16* Vp = Vt + (size_t)wh * CH * SPAD;

  auto issueK = [&](int jt_) {                  // 16x128 bf16 = 256 x b128
    int row = tid >> 4, c8 = (tid & 15) * 8;
    async_load_b128(Kp + (size_t)(jt_ * 16 + row) * CH + c8,
                    &ldsK[jt_ & 1][row * 128 + c8]);
  };
  auto issueV = [&](int p_) {                   // 128x32 bf16 = 512 x b128
#pragma unroll
    for (int u = 0; u < 2; ++u) {
      int c = tid + u * 256;
      int crow = c >> 2, j8 = (c & 3) * 8;
      async_load_b128(Vp + (size_t)crow * SPAD + p_ * 32 + j8,
                      &ldsV[p_ & 1][crow * 32 + j8]);
    }
  };

  // load Q rows into registers: 4 chunks of K=32
  v16bf aQ[4];
#pragma unroll
  for (int kc = 0; kc < 4; ++kc)
    aQ[kc] = make_a(Qp + (size_t)(qbase + n) * CH + kc * 32, g);

  v8f acc[8] = {};                              // 16 x 128 output accumulator
  float m_r[8], l_r[8], pPrev[8];
#pragma unroll
  for (int r = 0; r < 8; ++r) { m_r[r] = -__builtin_inff(); l_r[r] = 0.f; pPrev[r] = 0.f; }

  issueK(0);
  issueV(0);

  for (int jt = 0; jt < NJT; ++jt) {
    if (jt < NJT - 1) issueK(jt + 1);
    if ((jt & 1) && ((jt >> 1) + 1) < NJT / 2) issueV((jt >> 1) + 1);
    if (jt == NJT - 1) asm volatile("s_wait_asynccnt 0x0" ::: "memory");
    else               asm volatile("s_wait_asynccnt 0x3" ::: "memory");
    __syncthreads();                            // staged tiles visible to all

    v8f sc = {};
#pragma unroll
    for (int kc = 0; kc < 4; ++kc) {            // S = Q K^T over head dim 128
      v16bf bK = make_b(&ldsK[jt & 1][n * 128 + kc * 32], g);
      sc = wmma_bf16(aQ[kc], bK, sc);
    }
    int col = jt * 16 + n;
    float p[8], corr[8];
#pragma unroll
    for (int r = 0; r < 8; ++r) {
      float sv = sc[r] * SCALE;
      if (col >= SEQ) sv = -__builtin_inff();   // mask padded key columns
      float mt = rowmax16(sv);
      float mn = fmaxf(m_r[r], mt);
      corr[r]  = __expf(m_r[r] - mn);
      p[r]     = __expf(sv - mn);
      l_r[r]   = l_r[r] * corr[r] + rowsum16(p[r]);
      m_r[r]   = mn;
    }
#pragma unroll
    for (int nt = 0; nt < 8; ++nt)
#pragma unroll
      for (int r = 0; r < 8; ++r)
        acc[nt][r] *= corr[r];

    if ((jt & 1) == 0) {
#pragma unroll
      for (int r = 0; r < 8; ++r) pPrev[r] = p[r];
    } else {
      // write P pair (even tile rescaled to current max) into LDS,
      // read back in A-operand layout, then P*V^T WMMAs with K=32.
#pragma unroll
      for (int r = 0; r < 8; ++r) {
        ldsP[wave][0][(r + 8 * g) * 16 + n] = (__bf16)(pPrev[r] * corr[r]);
        ldsP[wave][1][(r + 8 * g) * 16 + n] = (__bf16)p[r];
      }
      __syncthreads();
      v16bf aP = {};
      {
        v8bf lo = *(const v8bf*)&ldsP[wave][0][n * 16 + 8 * g];
        v8bf hi = *(const v8bf*)&ldsP[wave][1][n * 16 + 8 * g];
#pragma unroll
        for (int h = 0; h < 8; ++h) { aP[h] = lo[h]; aP[h + 8] = hi[h]; }
      }
#pragma unroll
      for (int nt = 0; nt < 8; ++nt) {
        v16bf bV = make_b(&ldsV[(jt >> 1) & 1][(nt * 16 + n) * 32], g);
        acc[nt] = wmma_bf16(aP, bV, acc[nt]);
      }
    }
    __syncthreads();                            // safe to overwrite buffers
  }

#pragma unroll
  for (int nt = 0; nt < 8; ++nt)
#pragma unroll
    for (int r = 0; r < 8; ++r) {
      int srow = qbase + r + 8 * g;
      float o = acc[nt][r] / l_r[r];
      AO[(size_t)(wh * SPAD + srow) * CH + nt * 16 + n] = (__bf16)o;
    }
}

// ------------------------------------------- kernel 3: output projection ---
// grid (NTOK/128, DIM/64). Gathers attention output back to token order.
__global__ __launch_bounds__(256) void k_proj(
    const __bf16* __restrict__ AO, const float* __restrict__ Wp,
    const float* __restrict__ bp, float* __restrict__ out)
{
  __shared__ __bf16 ldsA[128 * 32];
  __shared__ __bf16 ldsW[64 * 32];
  const int tid  = threadIdx.x;
  const int wave = tid >> 5, lane = tid & 31;
  const int g = lane >> 4, n = lane & 15;
  const int tok0 = blockIdx.x * 128;
  const int d0   = blockIdx.y * 64;

  // precompute gather bases for this thread's two 8-element chunks per k-step
  int rws[2], subs[2];
  size_t base0[2];
#pragma unroll
  for (int u = 0; u < 2; ++u) {
    int c = tid + u * 256;                      // 512 v8bf chunks per k-step
    rws[u]  = c >> 2;
    subs[u] = (c & 3) * 8;
    int win, s; tok2win(tok0 + rws[u], win, s);
    base0[u] = (size_t)win * (NHEAD * SPAD * CH) + (size_t)s * CH;
  }

  v8f acc[4] = {};
  for (int ks = 0; ks < DIM / 32; ++ks) {
#pragma unroll
    for (int u = 0; u < 2; ++u) {               // gather AO rows (bf16, 16B)
      int cg = ks * 32 + subs[u];
      int head = cg >> 7, cc = cg & 127;
      v8bf v = *(const v8bf*)&AO[base0[u] + (size_t)head * (SPAD * CH) + cc];
      *(v8bf*)&ldsA[rws[u] * 32 + subs[u]] = v;
    }
#pragma unroll
    for (int i = 0; i < 2; ++i) {               // 64x32 weight tile
      int c = i * 256 + tid;
      int r = c >> 3, k4 = (c & 7) * 4;
      v4f f = *(const v4f*)&Wp[(size_t)(d0 + r) * DIM + ks * 32 + k4];
      v4bf b;
#pragma unroll
      for (int h = 0; h < 4; ++h) b[h] = (__bf16)f[h];
      *(v4bf*)&ldsW[r * 32 + k4] = b;
    }
    __syncthreads();
    v16bf a = make_a(&ldsA[(wave * 16 + n) * 32], g);
#pragma unroll
    for (int nt = 0; nt < 4; ++nt) {
      v16bf b = make_b(&ldsW[(nt * 16 + n) * 32], g);
      acc[nt] = wmma_bf16(a, b, acc[nt]);
    }
    __syncthreads();
  }

#pragma unroll
  for (int nt = 0; nt < 4; ++nt) {
    int d = d0 + nt * 16 + n;
    float bb = bp[d];
#pragma unroll
    for (int r = 0; r < 8; ++r) {
      int token = tok0 + wave * 16 + r + 8 * g;
      out[(size_t)token * DIM + d] = acc[nt][r] + bb;
    }
  }
}

// ------------------------------------------------------------------ launch ---
extern "C" void kernel_launch(void* const* d_in, const int* in_sizes, int n_in,
                              void* d_out, int out_size, void* d_ws, size_t ws_size,
                              hipStream_t stream) {
  (void)in_sizes; (void)n_in; (void)out_size; (void)ws_size;
  const float* x  = (const float*)d_in[0];
  const float* Wq = (const float*)d_in[1];
  const float* bq = (const float*)d_in[2];
  const float* Wk = (const float*)d_in[3];
  const float* bk = (const float*)d_in[4];
  const float* Wv = (const float*)d_in[5];
  const float* bv = (const float*)d_in[6];
  const float* Wp = (const float*)d_in[7];
  const float* bp = (const float*)d_in[8];

  const size_t tsz = (size_t)NWIN * NHEAD * SPAD * CH;   // elements per tensor
  __bf16* Qw = (__bf16*)d_ws;
  __bf16* Kw = Qw + tsz;
  __bf16* Vt = Kw + tsz;
  __bf16* AO = Vt + tsz;

  k_qkv <<<dim3(NTOK / 128, DIM / 64, 3), 256, 0, stream>>>(
      x, Wq, bq, Wk, bk, Wv, bv, Qw, Kw, Vt);
  k_attn<<<dim3(NWIN * NHEAD, SPAD / 128), 256, 0, stream>>>(Qw, Kw, Vt, AO);
  k_proj<<<dim3(NTOK / 128, DIM / 64), 256, 0, stream>>>(AO, Wp, bp, (float*)d_out);
}
